// GGN_IO_35983236006287
// MI455X (gfx1250) — matrix operand order repaired and checked by reading, not compiled, not run on hardware
//
#include <hip/hip_runtime.h>
#include <stdint.h>

#define N_NODES 100000
#define DIM     128
#define HID     256
#define PITCH   264      // padded bf16 row pitch (528B) -> consecutive K rows hit different LDS banks
#define NWAVES  8
#define BLOCK   256

typedef __attribute__((ext_vector_type(16))) __bf16 v16bf;
typedef __attribute__((ext_vector_type(8)))  float  v8f;

union BF16x16 {
  v16bf v;
  unsigned short s[16];
  uint4 q[2];
};

static __device__ __forceinline__ unsigned short f2bf(float f) {
  unsigned int u = __float_as_uint(f);
  unsigned int r = u + 0x7FFFu + ((u >> 16) & 1u);   // round-to-nearest-even
  return (unsigned short)(r >> 16);
}

// ---------------------------------------------------------------------------
// ws layout (bytes):
//   [0,      1024)   s_acc : float[256]
//   [1024,   2048)   c1    : float[256]   (b_n2e + Wr @ xi)
//   [2048,   69632)  W1t   : bf16[128][PITCH]  (W_n2e left half, transposed)
//   [69632, 204800)  W2t   : bf16[256][PITCH]  (W_e2e, transposed)
// ---------------------------------------------------------------------------

__global__ __launch_bounds__(BLOCK) void ggn_prep(
    const float* __restrict__ x, const int* __restrict__ ip,
    const float* __restrict__ W_n2e, const float* __restrict__ b_n2e,
    const float* __restrict__ W_e2e,
    float* __restrict__ s_acc, float* __restrict__ c1,
    unsigned short* __restrict__ W1t, unsigned short* __restrict__ W2t)
{
  int tid = threadIdx.x;
  int gid = blockIdx.x * BLOCK + tid;
  int gsz = gridDim.x * BLOCK;

  if (blockIdx.x == 0) {
    s_acc[tid] = 0.0f;
    int i = *ip;
    float acc = b_n2e[tid];
    for (int k = 0; k < DIM; ++k)
      acc += W_n2e[tid * (2 * DIM) + DIM + k] * x[(size_t)i * DIM + k];
    c1[tid] = acc;
  }
  for (int e = gid; e < DIM * PITCH; e += gsz) {
    int k = e / PITCH, n = e % PITCH;
    W1t[e] = (n < HID) ? f2bf(W_n2e[n * (2 * DIM) + k]) : (unsigned short)0;
  }
  for (int e = gid; e < HID * PITCH; e += gsz) {
    int k = e / PITCH, n = e % PITCH;
    W2t[e] = (n < HID) ? f2bf(W_e2e[n * HID + k]) : (unsigned short)0;
  }
}

// ---------------------------------------------------------------------------
// Fused: H1 = relu(x @ Wl^T + c1); H2 = relu(H1 @ W2^T + b_e2e);
//        s_acc += sum_rows adj[row] * H2[row, :]
// One wave = TWO 16-row strips (M-blocking): GEMM2 B fragments feed 2 WMMAs,
// halving GEMM2's LDS B traffic. Per-wave H1 LDS buffer reused serially;
// both strips' A2 fragments are held in registers for the dual-strip GEMM2.
// ---------------------------------------------------------------------------
__global__ __launch_bounds__(BLOCK) void ggn_main(
    const float* __restrict__ x, const float* __restrict__ adj,
    const float* __restrict__ c1, const float* __restrict__ b_e2e,
    const unsigned short* __restrict__ W1t,   // contiguous with W2t in ws
    float* __restrict__ s_acc)
{
  extern __shared__ char smem[];
  unsigned short* smW1  = (unsigned short*)smem;              // [128][PITCH]
  unsigned short* smW2  = smW1 + DIM * PITCH;                 // [256][PITCH]
  unsigned short* smH1a = smW1 + (DIM + HID) * PITCH;         // [NWAVES][16][PITCH]
  float*          smS   = (float*)(smH1a + NWAVES * 16 * PITCH);
  float*          smC1  = smS + HID;
  float*          smB2  = smC1 + HID;

  int tid = threadIdx.x;

  // stage W1t||W2t (contiguous 202752 B) into LDS with 16B vector copies
  {
    const uint4* src = (const uint4*)W1t;
    uint4* dst = (uint4*)smW1;
    const int n16 = (DIM + HID) * PITCH * 2 / 16;   // 12672
    for (int t = tid; t < n16; t += BLOCK) dst[t] = src[t];
  }
  smS[tid] = 0.0f;
  smC1[tid] = c1[tid];
  smB2[tid] = b_e2e[tid];
  __syncthreads();

  const int wave = tid >> 5;
  const int lane = tid & 31;
  const int half = lane >> 4;      // 0: lanes 0-15, 1: lanes 16-31
  const int l15  = lane & 15;
  unsigned short* myH1 = smH1a + wave * 16 * PITCH;

  const int pair = blockIdx.x * NWAVES + wave;     // pair of strips per wave
  const v8f vzero = {0.f,0.f,0.f,0.f,0.f,0.f,0.f,0.f};

  float aw[2][8];          // adjacency weights, per strip, per accumulator row
  v16bf a2[2][8];          // A2 fragments for both strips (held in registers)

  #pragma unroll
  for (int s = 0; s < 2; ++s) {
    const int rowBase = (pair * 2 + s) * 16;

    #pragma unroll
    for (int r = 0; r < 8; ++r) {
      int row = rowBase + half * 8 + r;
      aw[s][r] = (row < N_NODES) ? adj[row] : 0.0f;
    }

    // ---- A1 fragments: x rows, f32 -> bf16, ISA 16-bit A layout ----
    v16bf a1[4];
    {
      int row = rowBase + l15;
      if (row >= N_NODES) row = N_NODES - 1;   // clamped; contributes 0 via aw
      const float* rp0 = x + (size_t)row * DIM;
      #pragma unroll
      for (int kc = 0; kc < 4; ++kc) {
        const float* rp = rp0 + kc * 32 + half * 8;
        float4 f0 = ((const float4*)rp)[0];
        float4 f1 = ((const float4*)rp)[1];
        float4 f2 = ((const float4*)(rp + 16))[0];
        float4 f3 = ((const float4*)(rp + 16))[1];
        BF16x16 t;
        t.s[0]=f2bf(f0.x); t.s[1]=f2bf(f0.y); t.s[2]=f2bf(f0.z); t.s[3]=f2bf(f0.w);
        t.s[4]=f2bf(f1.x); t.s[5]=f2bf(f1.y); t.s[6]=f2bf(f1.z); t.s[7]=f2bf(f1.w);
        t.s[8]=f2bf(f2.x); t.s[9]=f2bf(f2.y); t.s[10]=f2bf(f2.z); t.s[11]=f2bf(f2.w);
        t.s[12]=f2bf(f3.x); t.s[13]=f2bf(f3.y); t.s[14]=f2bf(f3.z); t.s[15]=f2bf(f3.w);
        a1[kc] = t.v;
      }
    }

    // ---- GEMM1 (this strip): quad-N accumulators; relu; stage H1 in LDS ----
    for (int nt4 = 0; nt4 < 16; nt4 += 4) {
      v8f acc[4];
      #pragma unroll
      for (int j = 0; j < 4; ++j) acc[j] = vzero;
      #pragma unroll
      for (int kc = 0; kc < 4; ++kc) {
        const int krow = kc * 32 + half * 8;   // B: lane holds column n=l15, K in table order
        const unsigned short* bbase = smW1 + krow * PITCH + l15;
        #pragma unroll
        for (int j = 0; j < 4; ++j) {
          BF16x16 b;
          const unsigned short* bp = bbase + (nt4 + j) * 16;
          #pragma unroll
          for (int i = 0; i < 16; ++i) b.s[i] = bp[(i < 8 ? i : i + 8) * PITCH];
          acc[j] = __builtin_amdgcn_wmma_f32_16x16x32_bf16(
                       false, a1[kc], false, b.v, (short)0, acc[j], false, false);
        }
      }
      #pragma unroll
      for (int j = 0; j < 4; ++j) {
        float c1v = smC1[(nt4 + j) * 16 + l15];
        #pragma unroll
        for (int r = 0; r < 8; ++r) {               // D: vgpr r -> row m = half*8+r
          float h = fmaxf(acc[j][r] + c1v, 0.0f);
          myH1[(half * 8 + r) * PITCH + (nt4 + j) * 16 + l15] = f2bf(h);
        }
      }
    }

    // ---- A2 fragments for this strip (A layout); buffer is then reusable
    //      (same-wave DS ops are in-order) ----
    #pragma unroll
    for (int kc = 0; kc < 8; ++kc) {
      const unsigned short* hp = myH1 + l15 * PITCH + kc * 32 + half * 8;
      BF16x16 t;
      t.q[0] = ((const uint4*)hp)[0];
      t.q[1] = ((const uint4*)(hp + 16))[0];
      a2[s][kc] = t.v;
    }
  }

  // ---- GEMM2: dual-strip x dual-N blocking; each B fragment feeds 2 WMMAs ----
  for (int nt2 = 0; nt2 < 16; nt2 += 2) {
    v8f acc[2][2];
    acc[0][0] = vzero; acc[0][1] = vzero;
    acc[1][0] = vzero; acc[1][1] = vzero;
    #pragma unroll
    for (int kc = 0; kc < 8; ++kc) {
      const int krow = kc * 32 + half * 8;
      const unsigned short* bbase = smW2 + krow * PITCH + l15;
      BF16x16 b0, b1;
      const unsigned short* bp0 = bbase + nt2 * 16;
      const unsigned short* bp1 = bbase + (nt2 + 1) * 16;
      #pragma unroll
      for (int i = 0; i < 16; ++i) {
        b0.s[i] = bp0[(i < 8 ? i : i + 8) * PITCH];
        b1.s[i] = bp1[(i < 8 ? i : i + 8) * PITCH];
      }
      // 4 independent chains: dependent WMMAs stay 4 apart
      acc[0][0] = __builtin_amdgcn_wmma_f32_16x16x32_bf16(
                      false, a2[0][kc], false, b0.v, (short)0, acc[0][0], false, false);
      acc[1][0] = __builtin_amdgcn_wmma_f32_16x16x32_bf16(
                      false, a2[1][kc], false, b0.v, (short)0, acc[1][0], false, false);
      acc[0][1] = __builtin_amdgcn_wmma_f32_16x16x32_bf16(
                      false, a2[0][kc], false, b1.v, (short)0, acc[0][1], false, false);
      acc[1][1] = __builtin_amdgcn_wmma_f32_16x16x32_bf16(
                      false, a2[1][kc], false, b1.v, (short)0, acc[1][1], false, false);
    }
    #pragma unroll
    for (int j = 0; j < 2; ++j) {
      float b2v = smB2[(nt2 + j) * 16 + l15];
      float p = 0.0f;
      #pragma unroll
      for (int r = 0; r < 8; ++r) {
        p += aw[0][r] * fmaxf(acc[0][j][r] + b2v, 0.0f);
        p += aw[1][r] * fmaxf(acc[1][j][r] + b2v, 0.0f);
      }
      atomicAdd(&smS[(nt2 + j) * 16 + l15], p);     // ds_add_f32
    }
  }
  __syncthreads();
  atomicAdd(&s_acc[tid], smS[tid]);                 // one global atomic/col/block
}

// ---------------------------------------------------------------------------
// Tail: h = W_e2n @ s + b_e2n ; out = xi + concat(xi,h) @ W_out^T + b_out
// ---------------------------------------------------------------------------
__global__ __launch_bounds__(BLOCK) void ggn_final(
    const float* __restrict__ x, const int* __restrict__ ip,
    const float* __restrict__ W_e2n, const float* __restrict__ b_e2n,
    const float* __restrict__ W_out, const float* __restrict__ b_out,
    const float* __restrict__ s_acc, float* __restrict__ out)
{
  __shared__ float sS[HID];
  __shared__ float sh2[HID];
  int t = threadIdx.x;
  sS[t] = s_acc[t];
  __syncthreads();
  float h = b_e2n[t];
  for (int j = 0; j < HID; ++j) h += W_e2n[t * HID + j] * sS[j];
  sh2[t] = h;
  __syncthreads();
  if (t < DIM) {
    int i = *ip;
    const float* xi = x + (size_t)i * DIM;
    float acc = b_out[t];
    const float* wr = W_out + (size_t)t * (DIM + HID);
    for (int j = 0; j < DIM; ++j) acc += wr[j] * xi[j];
    for (int j = 0; j < HID; ++j) acc += wr[DIM + j] * sh2[j];
    out[t] = xi[t] + acc;
  }
}

extern "C" void kernel_launch(void* const* d_in, const int* in_sizes, int n_in,
                              void* d_out, int out_size, void* d_ws, size_t ws_size,
                              hipStream_t stream) {
  const float* x     = (const float*)d_in[0];
  const float* adj   = (const float*)d_in[1];
  const int*   ip    = (const int*)d_in[2];
  const float* W_n2e = (const float*)d_in[3];
  const float* b_n2e = (const float*)d_in[4];
  const float* W_e2e = (const float*)d_in[5];
  const float* b_e2e = (const float*)d_in[6];
  const float* W_e2n = (const float*)d_in[7];
  const float* b_e2n = (const float*)d_in[8];
  const float* W_out = (const float*)d_in[9];
  const float* b_out = (const float*)d_in[10];
  float* out = (float*)d_out;

  char* ws = (char*)d_ws;
  float* s_acc        = (float*)(ws);
  float* c1           = (float*)(ws + 1024);
  unsigned short* W1t = (unsigned short*)(ws + 2048);
  unsigned short* W2t = (unsigned short*)(ws + 69632);

  const int smemBytes = (DIM + HID) * PITCH * 2        // weights bf16
                      + NWAVES * 16 * PITCH * 2        // per-wave H1 staging
                      + HID * 4                        // block partial sums
                      + HID * 4                        // staged c1
                      + HID * 4;                       // staged b_e2e
  hipFuncSetAttribute((const void*)ggn_main,
                      hipFuncAttributeMaxDynamicSharedMemorySize, smemBytes);

  ggn_prep<<<64, BLOCK, 0, stream>>>(x, ip, W_n2e, b_n2e, W_e2e, s_acc, c1, W1t, W2t);

  const int strips = (N_NODES + 15) / 16;              // 6250
  const int pairs  = (strips + 1) / 2;                 // 3125
  const int blocks = (pairs + NWAVES - 1) / NWAVES;    // 391
  ggn_main<<<blocks, BLOCK, smemBytes, stream>>>(x, adj, c1, b_e2e, W1t, s_acc);

  ggn_final<<<1, BLOCK, 0, stream>>>(x, ip, W_e2n, b_e2n, W_out, b_out, s_acc, out);
}
